// MultiHeadAttention_25709674234172
// MI455X (gfx1250) — compile-verified, matrix operand order checked by reading, not run on hardware
//
#include <hip/hip_runtime.h>
#include <hip/hip_bf16.h>

// Problem constants (from reference)
#define BB 2
#define TT 2048
#define DD 1024
#define NH 16
#define HD 64

typedef __attribute__((ext_vector_type(16))) __bf16 v16bf;
typedef __attribute__((ext_vector_type(8)))  __bf16 v8bf;
typedef __attribute__((ext_vector_type(8)))  float  v8f;
typedef __attribute__((ext_vector_type(4)))  int    v4i;

typedef __attribute__((address_space(1))) v4i gv4i_t;   // global int4
typedef __attribute__((address_space(3))) v4i lv4i_t;   // LDS int4

// CDNA5 async global->LDS copy (ASYNCcnt-tracked). Falls back to a
// register-staged copy if the builtin is unavailable on this toolchain.
static __device__ __forceinline__ void async_copy_b128(const void* g, void* l) {
#if __has_builtin(__builtin_amdgcn_global_load_async_to_lds_b128)
    __builtin_amdgcn_global_load_async_to_lds_b128((gv4i_t*)g, (lv4i_t*)l, 0, 0);
#else
    *(v8bf*)l = *(const v8bf*)g;
#endif
}

template<int N>
static __device__ __forceinline__ void async_wait() {
#if __has_builtin(__builtin_amdgcn_s_wait_asynccnt)
    __builtin_amdgcn_s_wait_asynccnt(N);
#else
    asm volatile("s_wait_asynccnt %0" ::"n"(N) : "memory");
#endif
}

// Build a 16x32 (A-side) or 32x16 (B-side) bf16 WMMA fragment.
// Per lane: elements 0-7 from chunk0 (16B), elements 8-15 from chunk1 (16B).
// Caller passes base = row_ptr + 8*half ; second chunk at +16 elements.
static __device__ __forceinline__ v16bf make_frag(const __hip_bfloat16* p0) {
    union { v16bf v; v8bf h[2]; } u;
    u.h[0] = *(const v8bf*)(p0);
    u.h[1] = *(const v8bf*)(p0 + 16);
    return u.v;
}

// ---------------- conversion kernels ----------------
__global__ void cvt_bf16_kernel(const float* __restrict__ in,
                                __hip_bfloat16* __restrict__ out, int n) {
    int i = blockIdx.x * blockDim.x + threadIdx.x;
    if (i < n) out[i] = __float2bfloat16(in[i]);
}

// W[K][N] f32  ->  WT[N][K] bf16
__global__ void cvt_transpose_kernel(const float* __restrict__ W,
                                     __hip_bfloat16* __restrict__ WT,
                                     int K, int N) {
    int i = blockIdx.x * blockDim.x + threadIdx.x;
    if (i < K * N) {
        int k = i / N;
        int n = i - k * N;
        WT[(size_t)n * K + k] = __float2bfloat16(W[i]);
    }
}

// ---------------- WMMA GEMM (LDS double-buffered, async staged) ----------------
// C[M,N] = A[M,K] * B[K,N], B given transposed as BT[N][K] (both bf16).
// Block = 256 threads = 8 waves (2(M) x 4(N)); wave tile 64x64; block tile 128x256.
// Per K-step(32): async-stage 24KB into LDS (double buffered), 128 WMMA/block.
// mode 0: store f32 to Cout.
// mode 1: QKV scatter: Q,K as [B,H,T,HD] bf16; V stored transposed [B,H,HD,T] bf16.
#define BMT 128
#define BNT 256
#define BKT 32
#define LPAD 40   // padded LDS row (elements): bank-conflict-free ds_read_b128

__global__ void __launch_bounds__(256, 1)
gemm_wmma_kernel(const __hip_bfloat16* __restrict__ A,
                 const __hip_bfloat16* __restrict__ BT,
                 float* __restrict__ Cout,
                 __hip_bfloat16* __restrict__ qb,
                 __hip_bfloat16* __restrict__ kb,
                 __hip_bfloat16* __restrict__ vtb,
                 int M, int N, int K, int mode) {
    __shared__ __hip_bfloat16 Alds[2][BMT][LPAD];   // 20 KB
    __shared__ __hip_bfloat16 Blds[2][BNT][LPAD];   // 40 KB

    const int tid   = threadIdx.x;
    const int lane  = tid & 31;
    const int wave  = tid >> 5;
    const int laneM = lane & 15;
    const int half  = lane >> 4;
    const int mloc  = (wave >> 2) * 64;             // wave's M offset in block tile
    const int nloc  = (wave & 3) * 64;              // wave's N offset in block tile
    const int mblk  = blockIdx.x * BMT;
    const int nblk  = blockIdx.y * BNT;

    // stage one 128x32 A tile + 256x32 B tile: 6 x b128 per thread
    auto stage = [&](int buf, int k0) {
#pragma unroll
        for (int j = 0; j < 2; ++j) {               // A: 512 chunks
            const int c = tid + 256 * j;
            const int row = c >> 2, q = c & 3;
            async_copy_b128(A + (size_t)(mblk + row) * K + k0 + 8 * q,
                            &Alds[buf][row][8 * q]);
        }
#pragma unroll
        for (int j = 0; j < 4; ++j) {               // B: 1024 chunks
            const int c = tid + 256 * j;
            const int col = c >> 2, q = c & 3;
            async_copy_b128(BT + (size_t)(nblk + col) * K + k0 + 8 * q,
                            &Blds[buf][col][8 * q]);
        }
    };

    v8f acc[4][4];
#pragma unroll
    for (int mt = 0; mt < 4; ++mt)
#pragma unroll
        for (int nt = 0; nt < 4; ++nt) acc[mt][nt] = (v8f)(0.0f);

    stage(0, 0);
    const int nsteps = K / BKT;
    for (int i = 0; i < nsteps; ++i) {
        const int buf = i & 1;
        if (i + 1 < nsteps) {
            stage(buf ^ 1, (i + 1) * BKT);
            async_wait<6>();        // current buffer landed; 6 newest still in flight
        } else {
            async_wait<0>();
        }
        __syncthreads();            // everyone's tile data visible

        v16bf afr[4], bfr[4];
#pragma unroll
        for (int mt = 0; mt < 4; ++mt)
            afr[mt] = make_frag(&Alds[buf][mloc + 16 * mt + laneM][8 * half]);
#pragma unroll
        for (int nt = 0; nt < 4; ++nt)
            bfr[nt] = make_frag(&Blds[buf][nloc + 16 * nt + laneM][8 * half]);
#pragma unroll
        for (int mt = 0; mt < 4; ++mt)
#pragma unroll
            for (int nt = 0; nt < 4; ++nt)
                acc[mt][nt] = __builtin_amdgcn_wmma_f32_16x16x32_bf16(
                    false, afr[mt], false, bfr[nt], (short)0, acc[mt][nt], false, false);

        __syncthreads();            // all reads done before this buffer is refilled
    }

#pragma unroll
    for (int mt = 0; mt < 4; ++mt)
#pragma unroll
        for (int nt = 0; nt < 4; ++nt)
#pragma unroll
            for (int r = 0; r < 8; ++r) {
                const int row = mblk + mloc + 16 * mt + r + 8 * half;
                const int col = nblk + nloc + 16 * nt + laneM;
                const float v = acc[mt][nt][r];
                if (mode == 0) {
                    Cout[(size_t)row * N + col] = v;
                } else {
                    const int which = col >> 10;       // 0=q 1=k 2=v
                    const int d     = col & 1023;
                    const int h     = d >> 6;
                    const int hd    = d & 63;
                    const int b     = row >> 11;       // row / T
                    const int t     = row & 2047;
                    const __hip_bfloat16 bv = __float2bfloat16(v);
                    const size_t bh = (size_t)b * NH + h;
                    if (which == 0)       qb[(bh * TT + t) * HD + hd] = bv;
                    else if (which == 1)  kb[(bh * TT + t) * HD + hd] = bv;
                    else                  vtb[(bh * HD + hd) * TT + t] = bv;
                }
            }
}

// ---------------- causal flash attention ----------------
// Grid: B*H*(T/64) blocks, 128 threads (4 waves). Each wave owns 16 query rows.
// Q,K: [B,H,T,HD] bf16 ; Vt: [B,H,HD,T] bf16 ; out attnb: [B,T,D] bf16.
__global__ void __launch_bounds__(128, 1)
attn_kernel(const __hip_bfloat16* __restrict__ qb,
            const __hip_bfloat16* __restrict__ kb,
            const __hip_bfloat16* __restrict__ vtb,
            __hip_bfloat16* __restrict__ attnb) {
    __shared__ __hip_bfloat16 plds[4][16][32];   // per-wave P bounce tile

    const int lane  = threadIdx.x & 31;
    const int wave  = threadIdx.x >> 5;
    const int laneM = lane & 15;
    const int half  = lane >> 4;

    const int nqb  = TT / 64;                  // 32 query blocks
    const int qblk = blockIdx.x % nqb;
    const int bh   = blockIdx.x / nqb;         // b*H + h
    const int tq   = qblk * 64 + wave * 16;    // this wave's first query row

    const __hip_bfloat16* Qb = qb  + (size_t)bh * TT * HD;
    const __hip_bfloat16* Kb = kb  + (size_t)bh * TT * HD;
    const __hip_bfloat16* Vt = vtb + (size_t)bh * HD * TT;

    // Q fragments for rows tq..tq+15 (hd slices 0-31, 32-63): reused for all chunks.
    v16bf qf[2];
#pragma unroll
    for (int f = 0; f < 2; ++f)
        qf[f] = make_frag(Qb + (size_t)(tq + laneM) * HD + 32 * f + 8 * half);

    v8f o[4];
#pragma unroll
    for (int nt = 0; nt < 4; ++nt) o[nt] = (v8f)(0.0f);
    float mrow[8], lrow[8];
#pragma unroll
    for (int r = 0; r < 8; ++r) { mrow[r] = -INFINITY; lrow[r] = 0.0f; }

    const float sl2e = 0.125f * 1.4426950408889634f;   // HD^-0.5 * log2(e)
    const int nkt = (tq + 47) >> 5;                    // 32-wide key chunks over [0, tq+15]

    for (int kt = 0; kt < nkt; ++kt) {
        const int k0 = kt * 32;

        // hide latency: prefetch next chunk's K rows / V rows (global_prefetch_b8)
        if (kt + 1 < nkt) {
            __builtin_prefetch(Kb + (size_t)(k0 + 32 + lane) * HD, 0, 0);
            __builtin_prefetch(Vt + (size_t)(2 * lane) * TT + k0 + 32, 0, 0);
        }

        // S = Q * K^T for 16q x 32k (two 16-col subtiles, K-dim = HD -> 2 wmma each)
        v8f s[2];
        s[0] = (v8f)(0.0f); s[1] = (v8f)(0.0f);
#pragma unroll
        for (int ks = 0; ks < 2; ++ks)
#pragma unroll
            for (int f = 0; f < 2; ++f) {
                v16bf kf = make_frag(Kb + (size_t)(k0 + 16 * ks + laneM) * HD + 32 * f + 8 * half);
                s[ks] = __builtin_amdgcn_wmma_f32_16x16x32_bf16(
                    false, qf[f], false, kf, (short)0, s[ks], false, false);
            }

        // scale into log2 domain + causal mask (C layout: row=r+8*half, col=laneM)
        float sv[2][8];
#pragma unroll
        for (int ks = 0; ks < 2; ++ks)
#pragma unroll
            for (int r = 0; r < 8; ++r) {
                const int row = tq + r + 8 * half;
                const int col = k0 + 16 * ks + laneM;
                sv[ks][r] = (col <= row) ? s[ks][r] * sl2e : -INFINITY;
            }

        // online softmax: row max / exp2 / row sum / rescale O
        float pr[2][8];
#pragma unroll
        for (int r = 0; r < 8; ++r) {
            float mx = fmaxf(sv[0][r], sv[1][r]);
            mx = fmaxf(mx, __shfl_xor(mx, 1, 32));
            mx = fmaxf(mx, __shfl_xor(mx, 2, 32));
            mx = fmaxf(mx, __shfl_xor(mx, 4, 32));
            mx = fmaxf(mx, __shfl_xor(mx, 8, 32));
            const float mnew  = fmaxf(mrow[r], mx);
            const float alpha = exp2f(mrow[r] - mnew);
            const float p0 = exp2f(sv[0][r] - mnew);
            const float p1 = exp2f(sv[1][r] - mnew);
            pr[0][r] = p0; pr[1][r] = p1;
            float rsum = p0 + p1;
            rsum += __shfl_xor(rsum, 1, 32);
            rsum += __shfl_xor(rsum, 2, 32);
            rsum += __shfl_xor(rsum, 4, 32);
            rsum += __shfl_xor(rsum, 8, 32);
            lrow[r] = lrow[r] * alpha + rsum;
            mrow[r] = mnew;
#pragma unroll
            for (int nt = 0; nt < 4; ++nt) o[nt][r] *= alpha;
        }

        // Bounce P through LDS: C-layout -> A-layout (wave-private tile).
#pragma unroll
        for (int ks = 0; ks < 2; ++ks)
#pragma unroll
            for (int r = 0; r < 8; ++r)
                plds[wave][r + 8 * half][16 * ks + laneM] = __float2bfloat16(pr[ks][r]);
        asm volatile("s_wait_dscnt 0x0" ::: "memory");  // CDNA5 split DS counter

        v16bf pf = make_frag(&plds[wave][laneM][8 * half]);

        // O += P * V   (B-frag from Vt, contiguous along t)
#pragma unroll
        for (int nt = 0; nt < 4; ++nt) {
            v16bf vf = make_frag(Vt + (size_t)(16 * nt + laneM) * TT + k0 + 8 * half);
            o[nt] = __builtin_amdgcn_wmma_f32_16x16x32_bf16(
                false, pf, false, vf, (short)0, o[nt], false, false);
        }
    }

    // finalize + store [B,T,D] bf16
    const int b = bh / NH;
    const int h = bh % NH;
#pragma unroll
    for (int nt = 0; nt < 4; ++nt)
#pragma unroll
        for (int r = 0; r < 8; ++r) {
            const float v = o[nt][r] / lrow[r];
            const int t   = tq + r + 8 * half;
            const int col = h * HD + 16 * nt + laneM;
            attnb[((size_t)b * TT + t) * DD + col] = __float2bfloat16(v);
        }
}

// ---------------- launcher ----------------
extern "C" void kernel_launch(void* const* d_in, const int* in_sizes, int n_in,
                              void* d_out, int out_size, void* d_ws, size_t ws_size,
                              hipStream_t stream) {
    const float* x     = (const float*)d_in[0];   // [B,T,D]
    const float* w_qkv = (const float*)d_in[1];   // [D,3D]
    const float* w_out = (const float*)d_in[2];   // [D,D]
    float* out = (float*)d_out;                   // [B,T,D]

    char* ws = (char*)d_ws;
    size_t cur = 0;
    auto alloc = [&](size_t bytes) { char* p = ws + cur; cur += (bytes + 255) & ~(size_t)255; return p; };

    const int nX    = BB * TT * DD;          // 4,194,304
    const int nWqkv = DD * 3 * DD;           // 3,145,728
    const int nWout = DD * DD;               // 1,048,576

    __hip_bfloat16* xb    = (__hip_bfloat16*)alloc((size_t)nX * 2);
    __hip_bfloat16* wqkvT = (__hip_bfloat16*)alloc((size_t)nWqkv * 2);
    __hip_bfloat16* woutT = (__hip_bfloat16*)alloc((size_t)nWout * 2);
    __hip_bfloat16* qb    = (__hip_bfloat16*)alloc((size_t)nX * 2);
    __hip_bfloat16* kb    = (__hip_bfloat16*)alloc((size_t)nX * 2);
    __hip_bfloat16* vtb   = (__hip_bfloat16*)alloc((size_t)nX * 2);
    __hip_bfloat16* attnb = (__hip_bfloat16*)alloc((size_t)nX * 2);

    // 1) conversions
    cvt_bf16_kernel<<<(nX + 255) / 256, 256, 0, stream>>>(x, xb, nX);
    cvt_transpose_kernel<<<(nWqkv + 255) / 256, 256, 0, stream>>>(w_qkv, wqkvT, DD, 3 * DD);
    cvt_transpose_kernel<<<(nWout + 255) / 256, 256, 0, stream>>>(w_out, woutT, DD, DD);

    // 2) QKV projection: [4096,1024] x [1024,3072] -> scatter Q/K/Vt (bf16)
    {
        dim3 grid((BB * TT) / BMT, (3 * DD) / BNT);
        gemm_wmma_kernel<<<grid, 256, 0, stream>>>(xb, wqkvT, nullptr, qb, kb, vtb,
                                                   BB * TT, 3 * DD, DD, 1);
    }

    // 3) causal flash attention -> attnb [B,T,D] bf16
    attn_kernel<<<BB * NH * (TT / 64), 128, 0, stream>>>(qb, kb, vtb, attnb);

    // 4) output projection: [4096,1024] x [1024,1024] -> f32 d_out
    {
        dim3 grid((BB * TT) / BMT, DD / BNT);
        gemm_wmma_kernel<<<grid, 256, 0, stream>>>(attnb, woutT, out, nullptr, nullptr, nullptr,
                                                   BB * TT, DD, DD, 0);
    }
    (void)in_sizes; (void)n_in; (void)out_size; (void)ws_size;
}
